// Attention_52012053954690
// MI455X (gfx1250) — matrix operand order repaired and checked by reading, not compiled, not run on hardware
//
#include <hip/hip_runtime.h>
#include <math.h>

typedef unsigned short u16;
typedef __attribute__((ext_vector_type(16))) __bf16 v16bf;
typedef __attribute__((ext_vector_type(8)))  float  v8f;
typedef __attribute__((ext_vector_type(8)))  unsigned short u16x8;

#define DEVINL __device__ __forceinline__

// Problem constants (fixed by the reference)
constexpr int Bn = 4;
constexpr int Cc = 256;
constexpr int Cq = 32;
constexpr int Nn = 24 * 24 * 24;   // 13824 queries
constexpr int Mm = 12 * 12 * 12;   // 1728 keys/values

// Workspace layout (units: u16/bf16 elements)
constexpr size_t OFF_WQ = 0;                                   // Wf|Wg|Wh bf16 [3][32][256]
constexpr size_t OFF_WV = OFF_WQ + (size_t)3 * Cq * Cc;        // Wv bf16 [256][32]
constexpr size_t OFF_Q  = OFF_WV + (size_t)Cc * Cq;            // Q  bf16 [B][N][Cq]
constexpr size_t OFF_GF = OFF_Q  + (size_t)Bn * Nn * Cq;       // g (pre-pool) [B][N][Cq]
constexpr size_t OFF_HF = OFF_GF + (size_t)Bn * Nn * Cq;       // h (pre-pool) [B][N][Cq]
constexpr size_t OFF_K  = OFF_HF + (size_t)Bn * Nn * Cq;       // K  bf16 [B][M][Cq]
constexpr size_t OFF_VT = OFF_K  + (size_t)Bn * Mm * Cq;       // Vt bf16 [B][Cq][M]

// ---------- helpers ----------
DEVINL u16 f2bf(float f) {                 // fp32 -> bf16 RNE (bit-level)
  unsigned u = __float_as_uint(f);
  unsigned r = u + 0x7FFFu + ((u >> 16) & 1u);
  return (u16)(r >> 16);
}
DEVINL float bf2f(u16 h) { return __uint_as_float(((unsigned)h) << 16); }

// Load a 16x32 bf16 WMMA operand fragment: this lane's 16 halves are two
// contiguous 16B chunks (A and B: K 16*(l>>4)..+15).
DEVINL v16bf load_pair(const u16* p0, const u16* p1) {
  union { u16x8 h[2]; v16bf v; } u;
  u.h[0] = *reinterpret_cast<const u16x8*>(p0);
  u.h[1] = *reinterpret_cast<const u16x8*>(p1);
  return u.v;
}

DEVINL v16bf ones_frag() {                 // all-ones bf16 B operand (row-sum trick)
  union { u16 h[16]; v16bf v; } u;
  #pragma unroll
  for (int i = 0; i < 16; ++i) u.h[i] = 0x3F80u;   // 1.0 in bf16
  return u.v;
}

DEVINL float redmax16(float v) {           // max over 16-lane group (cols of C tile)
  #pragma unroll
  for (int m = 1; m < 16; m <<= 1) v = fmaxf(v, __shfl_xor(v, m, 32));
  return v;
}

DEVINL v8f wmma_bf16(v16bf a, v16bf b, v8f c) {
  return __builtin_amdgcn_wmma_f32_16x16x32_bf16(false, a, false, b, (short)0, c, false, false);
}

// ---------- kernel 0: convert weights fp32 -> bf16 into workspace ----------
__global__ void wconv_kernel(const float* __restrict__ Wf, const float* __restrict__ Wg,
                             const float* __restrict__ Wh, const float* __restrict__ Wv,
                             u16* __restrict__ ws) {
  const int tid = threadIdx.x;
  for (int i = tid; i < Cq * Cc; i += blockDim.x) {
    ws[OFF_WQ + 0 * Cq * Cc + i] = f2bf(Wf[i]);
    ws[OFF_WQ + 1 * Cq * Cc + i] = f2bf(Wg[i]);
    ws[OFF_WQ + 2 * Cq * Cc + i] = f2bf(Wh[i]);
  }
  for (int i = tid; i < Cc * Cq; i += blockDim.x)
    ws[OFF_WV + i] = f2bf(Wv[i]);       // Wv is [C][Cq] row-major: A-operand ready
}

// ---------- kernel 1: WMMA projections f = Wf x, g = Wg x, h = Wh x ----------
// Grid: B * (N/64) workgroups of 128 threads (4 waves, 16 spatial cols each).
__global__ __launch_bounds__(128)
void proj_kernel(const float* __restrict__ x, u16* ws) {
  __shared__ u16 Xs[64 * 264];          // x tile as bf16, [n(64)][c(256)], stride 264
  const int b   = blockIdx.x / (Nn / 64);
  const int n0  = (blockIdx.x % (Nn / 64)) * 64;
  const int tid = threadIdx.x;

  // cooperative fill: transpose [c][n] fp32 -> [n][c] bf16
  {
    const int nl = (tid & 15) * 4;
    const int cb = tid >> 4;            // 0..7
    for (int c0 = 0; c0 < Cc; c0 += 8) {
      const int c = c0 + cb;
      const float4 v = *reinterpret_cast<const float4*>(
          &x[(size_t)b * Cc * Nn + (size_t)c * Nn + n0 + nl]);
      Xs[(nl + 0) * 264 + c] = f2bf(v.x);
      Xs[(nl + 1) * 264 + c] = f2bf(v.y);
      Xs[(nl + 2) * 264 + c] = f2bf(v.z);
      Xs[(nl + 3) * 264 + c] = f2bf(v.w);
    }
  }
  __syncthreads();

  const int wv = tid >> 5, lane = tid & 31;
  const int lr = lane & 15, lh = lane >> 4;
  const v8f vzero = {0.f, 0.f, 0.f, 0.f, 0.f, 0.f, 0.f, 0.f};
  v8f acc[3][2];
  #pragma unroll
  for (int p = 0; p < 3; ++p) { acc[p][0] = vzero; acc[p][1] = vzero; }

  const u16* xrow = &Xs[(wv * 16 + lr) * 264];
  #pragma unroll
  for (int ks = 0; ks < 8; ++ks) {       // K = 256 in steps of 32
    const v16bf bfrag = load_pair(&xrow[ks * 32 + lh * 16], &xrow[ks * 32 + lh * 16 + 8]);
    #pragma unroll
    for (int p = 0; p < 3; ++p) {
      #pragma unroll
      for (int t = 0; t < 2; ++t) {      // Cq = 32 -> two 16-row A tiles
        const u16* wr = &ws[OFF_WQ + (size_t)p * Cq * Cc + (size_t)(t * 16 + lr) * Cc + ks * 32];
        const v16bf afrag = load_pair(&wr[16 * lh], &wr[16 * lh + 8]);
        acc[p][t] = wmma_bf16(afrag, bfrag, acc[p][t]);
      }
    }
  }

  // D tile: row = cq-local (reg + 8*half), col = n (lane&15). Store [B][N][Cq] bf16.
  const int n = n0 + wv * 16 + lr;
  u16* dst[3] = { ws + OFF_Q, ws + OFF_GF, ws + OFF_HF };
  #pragma unroll
  for (int p = 0; p < 3; ++p)
    #pragma unroll
    for (int t = 0; t < 2; ++t)
      #pragma unroll
      for (int r = 0; r < 8; ++r) {
        const int cq = t * 16 + r + 8 * lh;
        dst[p][((size_t)b * Nn + n) * Cq + cq] = f2bf(acc[p][t][r]);
      }
}

// ---------- kernel 2: 2x2x2 maxpool -> K [B][M][Cq], Vt [B][Cq][M] ----------
__global__ void pool_kernel(u16* ws) {
  const size_t i = (size_t)blockIdx.x * 256 + threadIdx.x;   // over B*M*Cq
  const int cq = i % Cq;
  const size_t t = i / Cq;
  const int m = t % Mm;
  const int b = t / Mm;
  const int pd = m / 144, rem = m % 144, ph = rem / 12, pw = rem % 12;
  float gmax = -INFINITY, hmax = -INFINITY;
  #pragma unroll
  for (int dz = 0; dz < 2; ++dz)
    #pragma unroll
    for (int dy = 0; dy < 2; ++dy)
      #pragma unroll
      for (int dx = 0; dx < 2; ++dx) {
        const int n = (2 * pd + dz) * 576 + (2 * ph + dy) * 24 + (2 * pw + dx);
        const size_t base = ((size_t)b * Nn + n) * Cq + cq;
        gmax = fmaxf(gmax, bf2f(ws[OFF_GF + base]));
        hmax = fmaxf(hmax, bf2f(ws[OFF_HF + base]));
      }
  ws[OFF_K  + ((size_t)b * Mm + m) * Cq + cq] = f2bf(gmax);
  ws[OFF_VT + ((size_t)b * Cq + cq) * Mm + m] = f2bf(hmax);
}

// ---------- kernel 3: flash attention + fused output projection ----------
// Grid: B * (N/64) workgroups of 128 threads; each wave owns 16 query rows.
// kv tile = 64: 4 S WMMAs -> one softmax update -> 2 row-sum WMMAs + 4 PV WMMAs.
__global__ __launch_bounds__(128)
void attn_kernel(const float* __restrict__ x, const float* __restrict__ gamma_p,
                 const u16* __restrict__ ws, float* __restrict__ out) {
  __shared__ u16 P[4 * 16 * 72];        // per-wave 16x64 bf16 staging (stride 72)
  const int b   = blockIdx.x / (Nn / 64);
  const int qt  = blockIdx.x % (Nn / 64);
  const int tid = threadIdx.x, wv = tid >> 5, lane = tid & 31;
  const int lr = lane & 15, lh = lane >> 4;
  const int q0 = qt * 64 + wv * 16;

  const u16* Qb = ws + OFF_Q  + (size_t)b * Nn * Cq;
  const u16* Kb = ws + OFF_K  + (size_t)b * Mm * Cq;
  const u16* Vb = ws + OFF_VT + (size_t)b * Cq * Mm;
  u16* Pw = &P[wv * 16 * 72];

  // Q A-fragment (16 rows x K=32), loaded once
  const u16* qrow = Qb + (size_t)(q0 + lr) * Cq;
  const v16bf aQ = load_pair(qrow + 16 * lh, qrow + 16 * lh + 8);
  const v16bf bONE = ones_frag();

  const v8f vzero = {0.f, 0.f, 0.f, 0.f, 0.f, 0.f, 0.f, 0.f};
  float mrow[8];
  #pragma unroll
  for (int r = 0; r < 8; ++r) mrow[r] = -INFINITY;
  v8f o0 = vzero, o1 = vzero;           // O accum: 16 rows x Cq=32 (two C tiles)
  v8f lacc = vzero;                     // row-sum accum (every column = row sum)

  for (int mb = 0; mb < Mm; mb += 64) {
    if (mb + 64 < Mm) {                 // emits global_prefetch_b8
      __builtin_prefetch(Kb + (size_t)(mb + 64 + lane) * Cq, 0, 1);
      __builtin_prefetch(Kb + (size_t)(mb + 96 + lane) * Cq, 0, 1);
    }

    // S = Q K^T for 64 kv columns (4 B fragments, K rows are [M][Cq])
    v8f s[4];
    #pragma unroll
    for (int j = 0; j < 4; ++j) {
      const u16* kr = Kb + (size_t)(mb + j * 16 + lr) * Cq + 16 * lh;
      s[j] = wmma_bf16(aQ, load_pair(kr, kr + 8), vzero);
    }

    // one online-softmax update per 64 kv; stats align with C-tile row layout
    #pragma unroll
    for (int r = 0; r < 8; ++r) {
      const float t01  = fmaxf(fmaxf(s[0][r], s[1][r]), fmaxf(s[2][r], s[3][r]));
      const float mnew = fmaxf(mrow[r], redmax16(t01));
      const float scale = __expf(mrow[r] - mnew);
      mrow[r] = mnew;
      o0[r] *= scale;  o1[r] *= scale;  lacc[r] *= scale;
      const int row = r + 8 * lh;       // transpose P into A layout via LDS
      #pragma unroll
      for (int j = 0; j < 4; ++j)
        Pw[row * 72 + j * 16 + lr] = f2bf(__expf(s[j][r] - mnew));
    }

    // P A-fragments (16 x 64 split into two K=32 fragments)
    const u16* prow = &Pw[lr * 72];
    const v16bf aP0 = load_pair(prow + 16 * lh, prow + 16 * lh + 8);
    const v16bf aP1 = load_pair(prow + 32 + 16 * lh, prow + 40 + 16 * lh);

    // row sums: lacc += P * ones  (replaces shuffle-based sum reduction)
    lacc = wmma_bf16(aP0, bONE, lacc);
    lacc = wmma_bf16(aP1, bONE, lacc);

    // O += P V   (B = Vt rows contiguous in m; two c-halves x two K chunks)
    const u16* v00 = Vb + (size_t)lr * Mm + mb + 16 * lh;
    const u16* v10 = Vb + (size_t)(16 + lr) * Mm + mb + 16 * lh;
    o0 = wmma_bf16(aP0, load_pair(v00, v00 + 8), o0);
    o0 = wmma_bf16(aP1, load_pair(v00 + 32, v00 + 40), o0);
    o1 = wmma_bf16(aP0, load_pair(v10, v10 + 8), o1);
    o1 = wmma_bf16(aP1, load_pair(v10 + 32, v10 + 40), o1);
  }

  // normalize, stage O as [nq][cq] bf16 in LDS for the projection B-operand
  #pragma unroll
  for (int r = 0; r < 8; ++r) {
    const float inv = 1.f / lacc[r];    // every column of lacc holds the row sum
    const int row = r + 8 * lh;
    Pw[row * 72 + lr]      = f2bf(o0[r] * inv);
    Pw[row * 72 + 16 + lr] = f2bf(o1[r] * inv);
  }
  const u16* orow = &Pw[lr * 72 + 16 * lh];
  const v16bf bO = load_pair(orow, orow + 8);   // B: k=cq (32), n=nq (16)

  // out = gamma * (Wv O) + x ; 16 c-tiles, K = Cq = 32 (one WMMA each)
  const float gamma = *gamma_p;
  const u16* Wvb = ws + OFF_WV;
  for (int cb = 0; cb < 16; ++cb) {
    const u16* wr = Wvb + (size_t)(cb * 16 + lr) * Cq;
    const v16bf aW = load_pair(wr + 16 * lh, wr + 16 * lh + 8);
    const v8f d = wmma_bf16(aW, bO, vzero);
    #pragma unroll
    for (int r = 0; r < 8; ++r) {
      const int cg = cb * 16 + r + 8 * lh;
      const size_t idx = ((size_t)b * Cc + cg) * Nn + q0 + lr;
      out[idx] = gamma * d[r] + x[idx];
    }
  }
}

// ---------- host launch ----------
extern "C" void kernel_launch(void* const* d_in, const int* in_sizes, int n_in,
                              void* d_out, int out_size, void* d_ws, size_t ws_size,
                              hipStream_t stream) {
  const float* x     = (const float*)d_in[0];
  const float* Wf    = (const float*)d_in[1];
  const float* Wg    = (const float*)d_in[2];
  const float* Wh    = (const float*)d_in[3];
  const float* Wv    = (const float*)d_in[4];
  const float* gamma = (const float*)d_in[5];
  u16*   ws  = (u16*)d_ws;
  float* out = (float*)d_out;

  hipLaunchKernelGGL(wconv_kernel, dim3(1), dim3(256), 0, stream, Wf, Wg, Wh, Wv, ws);
  hipLaunchKernelGGL(proj_kernel,  dim3(Bn * (Nn / 64)), dim3(128), 0, stream, x, ws);
  hipLaunchKernelGGL(pool_kernel,  dim3((Bn * Mm * Cq) / 256), dim3(256), 0, stream, ws);
  hipLaunchKernelGGL(attn_kernel,  dim3(Bn * (Nn / 64)), dim3(128), 0, stream, x, gamma, ws, out);
}
